// RegularAttentionLayer_42743514530669
// MI455X (gfx1250) — matrix operand order repaired and checked by reading, not compiled, not run on hardware
//
#include <hip/hip_runtime.h>

// MI455X / gfx1250, wave32. All matmuls via v_wmma_f32_16x16x32_f16.
// Pipeline: f32->f16 converts -> QKV WMMA GEMM -> flash-attention (online
// softmax, per-wave LDS re-layout of P) -> projection WMMA GEMM (f32 out).

typedef _Float16 half_t;
typedef __attribute__((ext_vector_type(16))) _Float16 v16h;
typedef __attribute__((ext_vector_type(8)))  _Float16 v8h;
typedef __attribute__((ext_vector_type(8)))  float    v8f;

#define B_   2
#define N_   4096
#define C_   512
#define H_   8
#define DH_  64
#define ROWS (B_ * N_)     // 8192
#define CQKV (3 * C_)      // 1536

__device__ __forceinline__ v8f wmma_f16(v16h a, v16h b, v8f c) {
  // D = A(16x32,f16) * B(32x16,f16) + C(16x16,f32)
  return __builtin_amdgcn_wmma_f32_16x16x32_f16(false, a, false, b, (short)0, c,
                                                false, false);
}

// A-fragment (16x32 f16) from a row-major matrix: rows row0..row0+15,
// cols k0..k0+31.  Lane l holds row (l&15),
// K = k0 + (l>>4)*16 + e  -> one contiguous 32B span (two 16B loads).
__device__ __forceinline__ v16h load_a16(const half_t* A, int ld, int row0, int k0) {
  const int lane = threadIdx.x & 31;
  const int m  = lane & 15;
  const int hh = lane >> 4;
  const half_t* p = A + (size_t)(row0 + m) * ld + (k0 + hh * 16);
  v8h lo = *(const v8h*)(p);
  v8h hi = *(const v8h*)(p + 8);
  v16h r;
#pragma unroll
  for (int e = 0; e < 8; ++e) { r[e] = lo[e]; r[e + 8] = hi[e]; }
  return r;
}

// B-fragment (32x16 f16) sourced from Bt = B^T stored row-major [N][K].
// Lane l holds column n=(l&15); K = k0 + (l>>4)*16 + e -> one 32B chunk.
__device__ __forceinline__ v16h load_b16(const half_t* Bt, int ld, int n0, int k0) {
  const int lane = threadIdx.x & 31;
  const int n = n0 + (lane & 15);
  const int k = k0 + ((lane >> 4) << 4);
  return *(const v16h*)(Bt + (size_t)n * ld + k);
}

// ---------------- precision converts ----------------

__global__ void f32_to_f16_kernel(const float* __restrict__ in,
                                  half_t* __restrict__ out, int n) {
  int i = blockIdx.x * blockDim.x + threadIdx.x;
  if (i < n) out[i] = (half_t)in[i];
}

// in: rows x cols (row-major) -> out: cols x rows (i.e. transposed), f16
__global__ void transpose_f16_kernel(const float* __restrict__ in,
                                     half_t* __restrict__ out, int rows, int cols) {
  int i = blockIdx.x * blockDim.x + threadIdx.x;
  if (i < rows * cols) {
    int r = i / cols, c = i % cols;
    out[(size_t)c * rows + r] = (half_t)in[i];
  }
}

// ---------------- QKV GEMM: [8192,512] @ [512,1536] + bias ----------------
// One wave per 16x16 output tile; writes Q,K row-major [bh][n][d], V transposed
// [bh][d][n] so the PV matmul gets contiguous B-fragments.

__global__ __launch_bounds__(256) void gemm_qkv_kernel(
    const half_t* __restrict__ X, const half_t* __restrict__ Wt,
    const float* __restrict__ bias, half_t* __restrict__ Q,
    half_t* __restrict__ K, half_t* __restrict__ Vt) {
  const int wave = blockIdx.x * 8 + (threadIdx.x >> 5);
  const int tm = wave / (CQKV / 16);
  const int tn = wave % (CQKV / 16);
  const int row0 = tm * 16, col0 = tn * 16;
  v8f c = {};
#pragma unroll 4
  for (int k0 = 0; k0 < C_; k0 += 32) {
    v16h a = load_a16(X, C_, row0, k0);
    v16h b = load_b16(Wt, C_, col0, k0);
    c = wmma_f16(a, b, c);
  }
  const int lane = threadIdx.x & 31;
  const int col  = col0 + (lane & 15);
  const float bv = bias[col];
  const int sect = col / C_;       // 0=q 1=k 2=v
  const int rem  = col % C_;
  const int h = rem / DH_, d = rem % DH_;
#pragma unroll
  for (int r = 0; r < 8; ++r) {
    const int rg = row0 + r + ((lane >> 4) << 3);   // C/D layout row
    const int bb = rg >> 12, n = rg & (N_ - 1);
    const half_t hv = (half_t)(c[r] + bv);
    const size_t bh = (size_t)(bb * H_ + h);
    if (sect == 0)      Q [(bh * N_ + n) * DH_ + d] = hv;
    else if (sect == 1) K [(bh * N_ + n) * DH_ + d] = hv;
    else                Vt[(bh * DH_ + d) * N_ + n] = hv;
  }
}

// ---------------- flash attention ----------------
// One wave per 16-query tile per (b,h). Streams 32-key blocks:
// 4 WMMA for S = (Q*scale) K^T, online softmax (row groups live in 16-lane
// halves per the D layout), P re-laid out through per-wave LDS, 4 WMMA for PV.

#define LDSS 40   // padded row stride (halves) to spread LDS banks

__global__ __launch_bounds__(256) void attn_kernel(
    const half_t* __restrict__ Q, const half_t* __restrict__ Km,
    const half_t* __restrict__ Vt, half_t* __restrict__ O) {
  __shared__ __align__(16) half_t plds[8][16 * LDSS];
  const int lane  = threadIdx.x & 31;
  const int wslot = threadIdx.x >> 5;
  const int wave  = blockIdx.x * 8 + wslot;
  const int qt = wave & (N_ / 16 - 1);
  const int bh = wave >> 8;
  const half_t* Qh = Q  + (size_t)bh * N_ * DH_;
  const half_t* Kh = Km + (size_t)bh * N_ * DH_;
  const half_t* Vh = Vt + (size_t)bh * DH_ * N_;
  half_t* lw = &plds[wslot][0];

  v16h q0 = load_a16(Qh, DH_, qt * 16, 0);
  v16h q1 = load_a16(Qh, DH_, qt * 16, 32);
  const _Float16 sc = (_Float16)0.125f;   // 1/sqrt(64), exact in f16
#pragma unroll
  for (int e = 0; e < 16; ++e) { q0[e] *= sc; q1[e] *= sc; }

  float mrow[8], lrow[8];
#pragma unroll
  for (int r = 0; r < 8; ++r) { mrow[r] = -3.0e38f; lrow[r] = 0.0f; }
  v8f a0 = {}, a1 = {}, a2 = {}, a3 = {};

  for (int j = 0; j < N_ / 32; ++j) {
    const int kb = j * 32;
    v16h b00 = load_b16(Kh, DH_, kb,      0);
    v16h b01 = load_b16(Kh, DH_, kb,      32);
    v16h b10 = load_b16(Kh, DH_, kb + 16, 0);
    v16h b11 = load_b16(Kh, DH_, kb + 16, 32);
    v8f s0 = {}, s1 = {};
    s0 = wmma_f16(q0, b00, s0);
    s0 = wmma_f16(q1, b01, s0);
    s1 = wmma_f16(q0, b10, s1);
    s1 = wmma_f16(q1, b11, s1);

#pragma unroll
    for (int r = 0; r < 8; ++r) {
      // row (r + 8*(lane>>4)), cols = lane&15 within each 16-key tile
      float t = fmaxf(s0[r], s1[r]);
#pragma unroll
      for (int d = 8; d >= 1; d >>= 1) t = fmaxf(t, __shfl_xor(t, d, 32));
      const float nm   = fmaxf(mrow[r], t);
      const float corr = __expf(mrow[r] - nm);
      mrow[r] = nm;
      const float p0 = __expf(s0[r] - nm);
      const float p1 = __expf(s1[r] - nm);
      float rs = p0 + p1;
#pragma unroll
      for (int d = 8; d >= 1; d >>= 1) rs += __shfl_xor(rs, d, 32);
      lrow[r] = lrow[r] * corr + rs;
      a0[r] *= corr; a1[r] *= corr; a2[r] *= corr; a3[r] *= corr;
      const int row = r + ((lane >> 4) << 3);
      lw[row * LDSS + (lane & 15)]      = (half_t)p0;
      lw[row * LDSS + 16 + (lane & 15)] = (half_t)p1;
    }
    // P (16x32) back in A-fragment layout via LDS (same wave, no barrier)
    v16h pa = load_a16(lw, LDSS, 0, 0);
    v16h v0 = load_b16(Vh, N_, 0,  kb);
    v16h v1 = load_b16(Vh, N_, 16, kb);
    v16h v2 = load_b16(Vh, N_, 32, kb);
    v16h v3 = load_b16(Vh, N_, 48, kb);
    a0 = wmma_f16(pa, v0, a0);
    a1 = wmma_f16(pa, v1, a1);
    a2 = wmma_f16(pa, v2, a2);
    a3 = wmma_f16(pa, v3, a3);
  }

  const int b = bh >> 3, h = bh & 7;
#pragma unroll
  for (int r = 0; r < 8; ++r) {
    const float inv = 1.0f / lrow[r];
    const int n = qt * 16 + r + ((lane >> 4) << 3);
    half_t* op = O + ((size_t)(b * N_ + n)) * C_ + h * DH_ + (lane & 15);
    op[0]  = (half_t)(a0[r] * inv);
    op[16] = (half_t)(a1[r] * inv);
    op[32] = (half_t)(a2[r] * inv);
    op[48] = (half_t)(a3[r] * inv);
  }
}

// ---------------- projection GEMM: [8192,512] @ [512,512] + bias -> f32 ----

__global__ __launch_bounds__(256) void gemm_proj_kernel(
    const half_t* __restrict__ A, const half_t* __restrict__ Wt,
    const float* __restrict__ bias, float* __restrict__ out) {
  const int wave = blockIdx.x * 8 + (threadIdx.x >> 5);
  const int tm = wave >> 5;       // 32 column tiles
  const int tn = wave & 31;
  const int row0 = tm * 16, col0 = tn * 16;
  v8f c = {};
#pragma unroll 4
  for (int k0 = 0; k0 < C_; k0 += 32) {
    v16h a = load_a16(A, C_, row0, k0);
    v16h b = load_b16(Wt, C_, col0, k0);
    c = wmma_f16(a, b, c);
  }
  const int lane = threadIdx.x & 31;
  const int col  = col0 + (lane & 15);
  const float bv = bias[col];
#pragma unroll
  for (int r = 0; r < 8; ++r) {
    const int rg = row0 + r + ((lane >> 4) << 3);
    out[(size_t)rg * C_ + col] = c[r] + bv;
  }
}

// ---------------- launch ----------------

extern "C" void kernel_launch(void* const* d_in, const int* in_sizes, int n_in,
                              void* d_out, int out_size, void* d_ws, size_t ws_size,
                              hipStream_t stream) {
  const float* x      = (const float*)d_in[0];
  const float* W_qkv  = (const float*)d_in[1];
  const float* b_qkv  = (const float*)d_in[2];
  const float* W_proj = (const float*)d_in[3];
  const float* b_proj = (const float*)d_in[4];
  float* out = (float*)d_out;

  // workspace layout (~44 MB total, 256B-aligned slices)
  char* ws = (char*)d_ws;
  size_t off = 0;
  auto take = [&](size_t bytes) {
    char* p = ws + off;
    off += (bytes + 255) & ~(size_t)255;
    return p;
  };
  half_t* X16  = (half_t*)take((size_t)ROWS * C_ * 2);    // x in f16
  half_t* WQT  = (half_t*)take((size_t)CQKV * C_ * 2);    // W_qkv^T f16 [1536][512]
  half_t* WPT  = (half_t*)take((size_t)C_ * C_ * 2);      // W_proj^T f16 [512][512]
  half_t* Qb   = (half_t*)take((size_t)B_ * H_ * N_ * DH_ * 2);
  half_t* Kb   = (half_t*)take((size_t)B_ * H_ * N_ * DH_ * 2);
  half_t* Vtb  = (half_t*)take((size_t)B_ * H_ * DH_ * N_ * 2);
  half_t* AO   = (half_t*)take((size_t)ROWS * C_ * 2);    // attention out f16
  (void)ws_size; (void)n_in; (void)in_sizes; (void)out_size;

  const int threads = 256;
  f32_to_f16_kernel<<<(ROWS * C_) / threads, threads, 0, stream>>>(x, X16, ROWS * C_);
  transpose_f16_kernel<<<(C_ * CQKV) / threads, threads, 0, stream>>>(W_qkv, WQT, C_, CQKV);
  transpose_f16_kernel<<<(C_ * C_) / threads, threads, 0, stream>>>(W_proj, WPT, C_, C_);

  // 512 x 96 tiles, 8 waves/block
  gemm_qkv_kernel<<<(ROWS / 16) * (CQKV / 16) / 8, threads, 0, stream>>>(
      X16, WQT, b_qkv, Qb, Kb, Vtb);

  // 16 bh * 256 qtiles = 4096 waves
  attn_kernel<<<(B_ * H_ * (N_ / 16)) / 8, threads, 0, stream>>>(Qb, Kb, Vtb, AO);

  // 512 x 32 tiles
  gemm_proj_kernel<<<(ROWS / 16) * (C_ / 16) / 8, threads, 0, stream>>>(
      AO, WPT, b_proj, out);
}
